// Classifier_76733885710748
// MI455X (gfx1250) — matrix operand order repaired and checked by reading, not compiled, not run on hardware
//
#include <hip/hip_runtime.h>

// CDNA5 / gfx1250 fused ViT-tiny classifier.
// 4 samples packed per 16-row WMMA tile, one tile per wave32, 8 waves/block.
// All GEMMs via v_wmma_f32_16x16x32_f16 (f16 inputs, f32 accumulate).
// Patchify K-axis is PERMUTED (K' = r*16 + c, c<14 real) so A-fragment runs are
// contiguous in x -> global_load_b64 with immediate offsets, no per-element div/mod.

typedef __attribute__((ext_vector_type(16))) _Float16 v16h;
typedef __attribute__((ext_vector_type(8)))  _Float16 v8h;
typedef __attribute__((ext_vector_type(8)))  float    v8f;
typedef __attribute__((ext_vector_type(4)))  float    v4f;
typedef __attribute__((ext_vector_type(2)))  float    f2;

#define WMMA(a, b, c) \
  __builtin_amdgcn_wmma_f32_16x16x32_f16(false, (a), false, (b), (short)0, (c), false, false)

static __device__ __forceinline__ v16h cat8(v8h lo, v8h hi) {
  return __builtin_shufflevector(lo, hi, 0,1,2,3,4,5,6,7,8,9,10,11,12,13,14,15);
}

// A-fragment from row-major [16][ldk] f16 matrix in LDS.
// Lane l: M = l&15; halfs 0-7: K = kbase + 8*hi + 0..7; halfs 8-15: K = kbase + 16 + 8*hi + 0..7
static __device__ __forceinline__ v16h ldA(const _Float16* m, int ldk, int kbase, int lane) {
  const int hi16 = (lane >> 4) & 1;
  const _Float16* p = m + (lane & 15) * ldk + kbase + 8 * hi16;
  v8h lo = *(const v8h*)p;
  v8h hi = *(const v8h*)(p + 16);
  return cat8(lo, hi);
}

// B-fragment from B-transposed row-major [N][ldk] f16 matrix in LDS (bt[n][k] = B[k][n]).
// Lane l: N = nbase + (l&15); halfs 0-7: K = kbase + 8*hi + 0..7; halfs 8-15: K = kbase + 16 + 8*hi + 0..7
static __device__ __forceinline__ v16h ldB(const _Float16* bt, int ldk, int nbase, int kbase, int lane) {
  const _Float16* p = bt + (nbase + (lane & 15)) * ldk + kbase + ((lane >> 4) << 3);
  v8h lo = *(const v8h*)p;
  v8h hi = *(const v8h*)(p + 16);
  return cat8(lo, hi);
}

// D-fragment (8 x f32) -> row-major f16 [16][ldn]. Lane l: N = nbase+(l&15), VGPR r: M = 8*hi + r
static __device__ __forceinline__ void stD16(_Float16* m, int ldn, int nbase, v8f d, int lane) {
  const int col = nbase + (lane & 15);
  const int rb  = (lane >> 4) << 3;
#pragma unroll
  for (int r = 0; r < 8; ++r) m[(rb + r) * ldn + col] = (_Float16)d[r];
}

// D-fragment -> row-major f32 [16][16]
static __device__ __forceinline__ void stD32(float* m, v8f d, int lane) {
  const int col = lane & 15;
  const int rb  = (lane >> 4) << 3;
#pragma unroll
  for (int r = 0; r < 8; ++r) m[(rb + r) * 16 + col] = d[r];
}

// 8 float2 quads for patchify K-step ks: run0 = x row 2ks (8 floats), run1 = x row 2ks+1.
// xq is the per-lane float2 pointer (already includes +8*hi16 floats). All offsets immediate.
static __device__ __forceinline__ void ldx(const f2* xq, int ks, f2 q[8]) {
#pragma unroll
  for (int i = 0; i < 4; ++i) q[i]     = xq[28 * ks + i];        // r = 2ks   (row*28 floats)
#pragma unroll
  for (int i = 0; i < 4; ++i) q[4 + i] = xq[28 * ks + 14 + i];   // r = 2ks+1
}

// Pack 16 floats -> A-fragment, zeroing padded K' positions (c=14,15 <=> quad i&3==3, hi16=1).
static __device__ __forceinline__ v16h packA(const f2 q[8], int hi16) {
  v16h a;
#pragma unroll
  for (int i = 0; i < 8; ++i) {
    const bool pad = (hi16 != 0) && ((i & 3) == 3);
    a[2 * i]     = (_Float16)(pad ? 0.0f : q[i][0]);
    a[2 * i + 1] = (_Float16)(pad ? 0.0f : q[i][1]);
  }
  return a;
}

#define N_ENC 6

__global__ __launch_bounds__(256) void vit_fused_kernel(
    const float* __restrict__ x,  const float* __restrict__ Wp,
    const float* __restrict__ Wq, const float* __restrict__ Wk,
    const float* __restrict__ Wv, const float* __restrict__ Wh,
    const float* __restrict__ Wc, const float* __restrict__ bc,
    float* __restrict__ out, int Btot)
{
  // LDS: 28672 + 2048 + 24576 + 8192 = 63488 bytes
  __shared__ __align__(16) _Float16 s_w[14336];        // WpT[64][224] (K' layout) / per-layer {WqT,WkT,WvT,WhT}
  __shared__ __align__(16) _Float16 s_wcT[1024];       // WcT[16][64] (rows 10-15 zero)
  __shared__ __align__(16) _Float16 s_scr[8][1536];    // per-wave scratch (h / q / k / v overlap)
  __shared__ __align__(16) float    s_S[8][256];       // per-wave 16x16 f32 score / logit tile

  const int tid  = threadIdx.x;
  const int wave = tid >> 5;
  const int lane = tid & 31;
  const int tileBase = blockIdx.x * 32 + wave * 4;

  _Float16* scr  = s_scr[wave];
  _Float16* hbuf = scr;          // 16x64 f16
  _Float16* qbuf = scr;          // 16x32 f16 (overlays hbuf[0:512], h is dead by then)
  _Float16* kbuf = scr + 512;    // 16x32 f16
  _Float16* vbuf = scr + 1024;   // 16x32 f16
  float*    Sw   = s_S[wave];    // 16x16 f32

  // ---- Stage WpT in permuted-K' layout (j' = r*16 + c, c<14 real) and WcT ----
  for (int i = tid; i < 64 * 224; i += 256) {              // 56 iters, uniform
    int n = i / 224, jp = i - n * 224;
    int r = jp >> 4, c = jp & 15;
    float w = Wp[(r * 14 + (c < 14 ? c : 0)) * 64 + n];
    s_w[i] = (_Float16)(c < 14 ? w : 0.0f);
  }
  for (int i = tid; i < 1024; i += 256) {
    int n = i >> 6, kk = i & 63;
    float w = Wc[kk * 10 + (n < 10 ? n : 0)];
    s_wcT[i] = (_Float16)(n < 10 ? w : 0.0f);
  }
  __syncthreads();

  const int Ml   = lane & 15;        // tile row: sample = Ml>>2, token = Ml&3
  const int hi16 = lane >> 4;
  int samp = tileBase + (Ml >> 2);
  if (samp >= Btot) samp = Btot - 1;                       // select, no branch (EXEC stays full)
  const int tok = Ml & 3;
  // patch (t) base: x[b, (t>>1)*14 + r, (t&1)*14 + c]; per-lane +8*hi16 floats for the K' runs
  const float* xb = x + (size_t)samp * 784 + (tok >> 1) * 392 + (tok & 1) * 14;
  const f2* xq = (const f2*)xb + 4 * hi16;                 // 8-byte aligned (all terms even floats)

  // ---- Patchify GEMM: h0[16x64] = p[16x224'] @ Wp'[224'x64], 7 K-steps, pipelined loads ----
  v8f acc0 = {}, acc1 = {}, acc2 = {}, acc3 = {};
  f2 cur[8], nxt[8];
  ldx(xq, 0, cur);
#pragma unroll
  for (int ks = 0; ks < 7; ++ks) {
    if (ks < 6) ldx(xq, ks + 1, nxt);                      // prefetch next step's 8 quads
    v16h a  = packA(cur, hi16);
    v16h b0 = ldB(s_w, 224,  0, 32 * ks, lane);
    v16h b1 = ldB(s_w, 224, 16, 32 * ks, lane);
    v16h b2 = ldB(s_w, 224, 32, 32 * ks, lane);
    v16h b3 = ldB(s_w, 224, 48, 32 * ks, lane);
    acc0 = WMMA(a, b0, acc0);
    acc1 = WMMA(a, b1, acc1);
    acc2 = WMMA(a, b2, acc2);
    acc3 = WMMA(a, b3, acc3);
#pragma unroll
    for (int i = 0; i < 8; ++i) cur[i] = nxt[i];
  }
  stD16(hbuf, 64,  0, acc0, lane);
  stD16(hbuf, 64, 16, acc1, lane);
  stD16(hbuf, 64, 32, acc2, lane);
  stD16(hbuf, 64, 48, acc3, lane);

  // ---- Encoder layers ----
  for (int L = 0; L < N_ENC; ++L) {
    __syncthreads();
    const float* wq = Wq + L * 64 * 24;
    const float* wk = Wk + L * 64 * 24;
    const float* wv = Wv + L * 64 * 32;
    const float* wh = Wh + L * 32 * 64;
    for (int i = tid; i < 2048; i += 256) {                // uniform, 8 iters
      int n = i >> 6, kk = i & 63;
      int nc = n < 24 ? n : 0;
      float q = wq[kk * 24 + nc];
      float k = wk[kk * 24 + nc];
      s_w[i]        = (_Float16)(n < 24 ? q : 0.0f);       // WqT[32][64] (rows 24-31 zero)
      s_w[2048 + i] = (_Float16)(n < 24 ? k : 0.0f);       // WkT[32][64]
      s_w[4096 + i] = (_Float16)wv[kk * 32 + n];           // WvT[32][64]
      int n2 = i >> 5, k2 = i & 31;
      s_w[6144 + i] = (_Float16)wh[k2 * 64 + n2];          // WhT[64][32]
    }
    __syncthreads();

    v16h a0 = ldA(hbuf, 64, 0,  lane);                     // h A-frags (K=64 -> 2 steps)
    v16h a1 = ldA(hbuf, 64, 32, lane);

#pragma unroll
    for (int nt = 0; nt < 2; ++nt) {
      // batch the 6 B-fragment loads, then 6 WMMAs back-to-back
      v16h bq0 = ldB(s_w,        64, nt * 16, 0,  lane);
      v16h bq1 = ldB(s_w,        64, nt * 16, 32, lane);
      v16h bk0 = ldB(s_w + 2048, 64, nt * 16, 0,  lane);
      v16h bk1 = ldB(s_w + 2048, 64, nt * 16, 32, lane);
      v16h bv0 = ldB(s_w + 4096, 64, nt * 16, 0,  lane);
      v16h bv1 = ldB(s_w + 4096, 64, nt * 16, 32, lane);
      v8f qd = {}; qd = WMMA(a0, bq0, qd); qd = WMMA(a1, bq1, qd);
      v8f kd = {}; kd = WMMA(a0, bk0, kd); kd = WMMA(a1, bk1, kd);
      v8f vd = {}; vd = WMMA(a0, bv0, vd); vd = WMMA(a1, bv1, vd);
      stD16(qbuf, 32, nt * 16, qd, lane);
      stD16(kbuf, 32, nt * 16, kd, lane);
      stD16(vbuf, 32, nt * 16, vd, lane);
    }

    // Scores: S[16][16] = q[16x32] @ k^T[32x16]; per-sample 4x4 blocks on the diagonal
    {
      v16h aq = ldA(qbuf, 32, 0, lane);
      v16h bk = ldB(kbuf, 32, 0, 0, lane);                 // k row-major == (k^T) in B-layout
      v8f sc = {};
      sc = WMMA(aq, bk, sc);
      stD32(Sw, sc, lane);
    }
    __builtin_amdgcn_wave_barrier();                       // cross-lane reads of Sw follow

    // Per-lane: softmax over QUERY axis (reference: softmax(a, axis=1)) for this lane's sample
    const int s4 = (Ml >> 2) << 2;                         // sample row base in tile
    const int qi = Ml & 3;
    float blk[4][4];
#pragma unroll
    for (int i = 0; i < 4; ++i) {
      v4f row = *(const v4f*)&Sw[(s4 + i) * 16 + s4];
      blk[i][0] = row[0]; blk[i][1] = row[1]; blk[i][2] = row[2]; blk[i][3] = row[3];
    }
    float arow[4];
#pragma unroll
    for (int kk = 0; kk < 4; ++kk) {
      float mx = fmaxf(fmaxf(blk[0][kk], blk[1][kk]), fmaxf(blk[2][kk], blk[3][kk]));
      float e0 = __expf(blk[0][kk] - mx), e1 = __expf(blk[1][kk] - mx);
      float e2 = __expf(blk[2][kk] - mx), e3 = __expf(blk[3][kk] - mx);
      float den = e0 + e1 + e2 + e3;
      float eq = (qi == 0) ? e0 : (qi == 1) ? e1 : (qi == 2) ? e2 : e3;
      arow[kk] = eq / den;
    }

    // attn_out[16x32] = a @ v, built directly in A-fragment layout (K-cols this lane owns)
    v16h af;
#pragma unroll
    for (int g = 0; g < 2; ++g) {
      int cb = g * 16 + 8 * hi16;
      float a8[8] = {0, 0, 0, 0, 0, 0, 0, 0};
#pragma unroll
      for (int kk = 0; kk < 4; ++kk) {
        v8h vrow = *(const v8h*)&vbuf[(s4 + kk) * 32 + cb];
#pragma unroll
        for (int jj = 0; jj < 8; ++jj) a8[jj] += arow[kk] * (float)vrow[jj];
      }
#pragma unroll
      for (int jj = 0; jj < 8; ++jj) af[g * 8 + jj] = (_Float16)a8[jj];
    }

    // h_new[16x64] = attn_out[16x32] @ Wh[32x64]  (K=32 -> single step)
    {
      v16h bh0 = ldB(s_w + 6144, 32,  0, 0, lane);
      v16h bh1 = ldB(s_w + 6144, 32, 16, 0, lane);
      v16h bh2 = ldB(s_w + 6144, 32, 32, 0, lane);
      v16h bh3 = ldB(s_w + 6144, 32, 48, 0, lane);
      v8f h0 = {}; h0 = WMMA(af, bh0, h0);
      v8f h1 = {}; h1 = WMMA(af, bh1, h1);
      v8f h2 = {}; h2 = WMMA(af, bh2, h2);
      v8f h3 = {}; h3 = WMMA(af, bh3, h3);
      stD16(hbuf, 64,  0, h0, lane);
      stD16(hbuf, 64, 16, h1, lane);
      stD16(hbuf, 64, 32, h2, lane);
      stD16(hbuf, 64, 48, h3, lane);
    }
  }

  // ---- Classifier: logits[16x16] = h[16x64] @ Wc[64x16(pad)], then token mean + bias ----
  {
    v16h a0 = ldA(hbuf, 64, 0,  lane);
    v16h a1 = ldA(hbuf, 64, 32, lane);
    v16h b0 = ldB(s_wcT, 64, 0, 0,  lane);
    v16h b1 = ldB(s_wcT, 64, 0, 32, lane);
    v8f lg = {};
    lg = WMMA(a0, b0, lg);
    lg = WMMA(a1, b1, lg);
    stD32(Sw, lg, lane);
  }
  __builtin_amdgcn_wave_barrier();
  if (lane < 10) {                                         // after the last WMMA: divergence OK
    float bias = bc[lane];
#pragma unroll
    for (int s = 0; s < 4; ++s) {
      int sg = tileBase + s;
      if (sg < Btot) {
        float v = 0.25f * (Sw[(4 * s + 0) * 16 + lane] + Sw[(4 * s + 1) * 16 + lane] +
                           Sw[(4 * s + 2) * 16 + lane] + Sw[(4 * s + 3) * 16 + lane]) + bias;
        out[(size_t)sg * 10 + lane] = v;
      }
    }
  }
}

extern "C" void kernel_launch(void* const* d_in, const int* in_sizes, int n_in,
                              void* d_out, int out_size, void* d_ws, size_t ws_size,
                              hipStream_t stream) {
  const float* x  = (const float*)d_in[0];
  const float* Wp = (const float*)d_in[1];
  const float* Wq = (const float*)d_in[2];
  const float* Wk = (const float*)d_in[3];
  const float* Wv = (const float*)d_in[4];
  const float* Wh = (const float*)d_in[5];
  const float* Wc = (const float*)d_in[6];
  const float* bc = (const float*)d_in[7];
  float* out = (float*)d_out;

  int Btot = in_sizes[0] / 784;                 // 32768
  int grid = (Btot + 31) / 32;                  // 32 samples per 256-thread block
  vit_fused_kernel<<<dim3(grid), dim3(256), 0, stream>>>(
      x, Wp, Wq, Wk, Wv, Wh, Wc, bc, out, Btot);
}
